// NPUDeepseekSparseAttention_61890478735735
// MI455X (gfx1250) — matrix operand order repaired and checked by reading, not compiled, not run on hardware
//
#include <hip/hip_runtime.h>

// ---------------------------------------------------------------------------
// DeepSeek sparse attention for MI455X (gfx1250): all GEMM-shaped work via
// v_wmma_f32_16x16x32_bf16, norms/rope/softmax/top-k in f32 VALU.
// GEMM: 32x64 tile per wave (2 A frags x 4 B frags = 8 WMMA per k-step)
// ---------------------------------------------------------------------------

typedef unsigned short u16;
typedef __bf16 bf16;
typedef __attribute__((ext_vector_type(16))) __bf16 v16bf;
typedef __attribute__((ext_vector_type(8)))  __bf16 v8bf;
typedef __attribute__((ext_vector_type(8)))  float  v8f;

#define T_    2048
#define HID_  2048
#define H_    16
#define NOPE_ 128
#define ROPE_ 64
#define V_    128
#define QLR_  1536
#define KVLR_ 512
#define HI_   16
#define IDXD_ 128
#define NEGF  (-1e30f)
#define ATT_SCALE 0.07216878364870323f   // (NOPE+ROPE)^-0.5
#define IDX_SCALE 0.08838834764831845f   // IDX_D^-0.5

__device__ __forceinline__ u16 f2bf(float x) {
  unsigned int u = __float_as_uint(x);
  unsigned int r = (u + 0x7fffu + ((u >> 16) & 1u)) >> 16;
  return (u16)r;
}

// A fragment: 16x32 bf16, row-major source [rows][K].
// lane 0-15 : row=lane,    K 0..7 (v0-3) and 16..23 (v4-7)
// lane 16-31: row=lane-16, K 8..15        and 24..31
__device__ __forceinline__ v16bf load_fragA(const u16* base, long ld, int r0, int k0) {
  int lane = threadIdx.x & 31;
  const bf16* p = (const bf16*)(base + (long)(r0 + (lane & 15)) * ld + k0 + ((lane >> 4) << 3));
  v8bf lo = *(const v8bf*)p;
  v8bf hi = *(const v8bf*)(p + 16);
  v16bf f;
#pragma unroll
  for (int i = 0; i < 8; ++i) { f[i] = lo[i]; f[i + 8] = hi[i]; }
  return f;
}

// B fragment from B^T stored row-major [N][K].
// lane 0-15: col n=lane holds K 0..7 and 16..23; lane 16-31: col n=lane-16 holds K 8..15 and 24..31
__device__ __forceinline__ v16bf load_fragB(const u16* base, long ld, int n0, int k0) {
  int lane = threadIdx.x & 31;
  const bf16* p = (const bf16*)(base + (long)(n0 + (lane & 15)) * ld + k0 + ((lane >> 4) << 3));
  v8bf lo = *(const v8bf*)p;
  v8bf hi = *(const v8bf*)(p + 16);
  v16bf f;
#pragma unroll
  for (int i = 0; i < 8; ++i) { f[i] = lo[i]; f[i + 8] = hi[i]; }
  return f;
}

__device__ __forceinline__ v8f wmma_bf16(v16bf a, v16bf b, v8f c) {
  return __builtin_amdgcn_wmma_f32_16x16x32_bf16(false, a, false, b, (short)0, c, false, false);
}

// ---------------------------------------------------------------------------
// Generic batched GEMM: C[b](MxN) = A[b](MxK) @ B[b]^T, B stored as (N,K).
// One wave -> 32x64 tile: 2 A frags, 4 B frags, 8 WMMAs per k-step of 32.
// Requires M%32==0, N%64==0, K%32==0 (true for all call sites here).
// ---------------------------------------------------------------------------
__global__ void gemm_wmma(const u16* __restrict__ A, long lda, long sA,
                          const u16* __restrict__ B, long ldb, long sB,
                          void* __restrict__ C, long ldc, long sC, int cbf,
                          int M, int N, int K, int batch) {
  int wave = (int)((blockIdx.x * blockDim.x + threadIdx.x) >> 5);
  int tilesN = N >> 6;
  int perB = (M >> 5) * tilesN;
  if (wave >= batch * perB) return;
  int b  = wave / perB;
  int rr = wave - b * perB;
  int tm = rr / tilesN;
  int tn = rr - tm * tilesN;
  int m0 = tm << 5, n0 = tn << 6;
  const u16* Ab = A + (long)b * sA;
  const u16* Bb = B + (long)b * sB;
  v8f acc00 = {}, acc01 = {}, acc02 = {}, acc03 = {};
  v8f acc10 = {}, acc11 = {}, acc12 = {}, acc13 = {};
  for (int k = 0; k < K; k += 32) {
    __builtin_prefetch((const void*)(Ab + (long)(m0 + (threadIdx.x & 15)) * lda + k + 32), 0, 1);
    v16bf a0 = load_fragA(Ab, lda, m0,      k);
    v16bf a1 = load_fragA(Ab, lda, m0 + 16, k);
    v16bf b0 = load_fragB(Bb, ldb, n0,      k);
    v16bf b1 = load_fragB(Bb, ldb, n0 + 16, k);
    v16bf b2 = load_fragB(Bb, ldb, n0 + 32, k);
    v16bf b3 = load_fragB(Bb, ldb, n0 + 48, k);
    acc00 = wmma_bf16(a0, b0, acc00);
    acc01 = wmma_bf16(a0, b1, acc01);
    acc02 = wmma_bf16(a0, b2, acc02);
    acc03 = wmma_bf16(a0, b3, acc03);
    acc10 = wmma_bf16(a1, b0, acc10);
    acc11 = wmma_bf16(a1, b1, acc11);
    acc12 = wmma_bf16(a1, b2, acc12);
    acc13 = wmma_bf16(a1, b3, acc13);
  }
  int lane = threadIdx.x & 31;
  int n = lane & 15, mh = (lane >> 4) << 3;
  if (cbf) {
    u16* Cb = (u16*)C + (long)b * sC;
#pragma unroll
    for (int r = 0; r < 8; ++r) {
      long row0 = (long)(m0 + mh + r) * ldc;
      long row1 = (long)(m0 + 16 + mh + r) * ldc;
      Cb[row0 + n0 + n]      = f2bf(acc00[r]);
      Cb[row0 + n0 + 16 + n] = f2bf(acc01[r]);
      Cb[row0 + n0 + 32 + n] = f2bf(acc02[r]);
      Cb[row0 + n0 + 48 + n] = f2bf(acc03[r]);
      Cb[row1 + n0 + n]      = f2bf(acc10[r]);
      Cb[row1 + n0 + 16 + n] = f2bf(acc11[r]);
      Cb[row1 + n0 + 32 + n] = f2bf(acc12[r]);
      Cb[row1 + n0 + 48 + n] = f2bf(acc13[r]);
    }
  } else {
    float* Cf = (float*)C + (long)b * sC;
#pragma unroll
    for (int r = 0; r < 8; ++r) {
      long row0 = (long)(m0 + mh + r) * ldc;
      long row1 = (long)(m0 + 16 + mh + r) * ldc;
      Cf[row0 + n0 + n]      = acc00[r];
      Cf[row0 + n0 + 16 + n] = acc01[r];
      Cf[row0 + n0 + 32 + n] = acc02[r];
      Cf[row0 + n0 + 48 + n] = acc03[r];
      Cf[row1 + n0 + n]      = acc10[r];
      Cf[row1 + n0 + 16 + n] = acc11[r];
      Cf[row1 + n0 + 32 + n] = acc12[r];
      Cf[row1 + n0 + 48 + n] = acc13[r];
    }
  }
}

// ---------------------------------------------------------------------------
// Elementwise / normalization kernels
// ---------------------------------------------------------------------------
__global__ void convert_bf16_k(const float* __restrict__ in, u16* __restrict__ out, long n) {
  long i = (long)blockIdx.x * blockDim.x + threadIdx.x;
  long stride = (long)gridDim.x * blockDim.x;
  for (; i < n; i += stride) out[i] = f2bf(in[i]);
}

// Wkv_b (H,256,512) -> W_UK^T (H,512,128) and W_UV (H,128,512), both bf16
__global__ void split_wkvb_k(const float* __restrict__ Wkv_b,
                             u16* __restrict__ wukt, u16* __restrict__ wuv) {
  const long per = (long)H_ * KVLR_ * NOPE_;
  long idx = (long)blockIdx.x * blockDim.x + threadIdx.x;
  if (idx >= 2 * per) return;
  if (idx < per) {
    int h  = (int)(idx / (KVLR_ * NOPE_));
    int rr = (int)(idx % (KVLR_ * NOPE_));
    int d = rr / NOPE_, n = rr % NOPE_;
    wukt[idx] = f2bf(Wkv_b[((long)(h * 256 + n)) * KVLR_ + d]);
  } else {
    long r = idx - per;
    int h  = (int)(r / (V_ * KVLR_));
    int rr = (int)(r % (V_ * KVLR_));
    int n = rr / KVLR_, d = rr % KVLR_;
    wuv[r] = f2bf(Wkv_b[((long)(h * 256 + 128 + n)) * KVLR_ + d]);
  }
}

__global__ void rmsnorm_rows_bf_k(const float* __restrict__ in, const float* __restrict__ g,
                                  u16* __restrict__ out, int ncol) {
  __shared__ float red[256];
  int t = blockIdx.x, tid = threadIdx.x;
  const float* row = in + (long)t * ncol;
  float ss = 0.f;
  for (int i = tid; i < ncol; i += 256) { float v = row[i]; ss += v * v; }
  red[tid] = ss; __syncthreads();
  for (int o = 128; o > 0; o >>= 1) { if (tid < o) red[tid] += red[tid + o]; __syncthreads(); }
  float inv = rsqrtf(red[0] / (float)ncol + 1e-6f);
  for (int i = tid; i < ncol; i += 256) out[(long)t * ncol + i] = f2bf(row[i] * inv * g[i]);
}

// q (T,H,192) f32 -> q_nope bf16 (T,H,128), roped q_pe bf16 (T,H,64)
__global__ void extract_q_k(const float* __restrict__ q, const float* __restrict__ cosb,
                            const float* __restrict__ sinb,
                            u16* __restrict__ qn, u16* __restrict__ qpe) {
  long idx = (long)blockIdx.x * blockDim.x + threadIdx.x;
  if (idx >= (long)T_ * H_ * 192) return;
  int t = (int)(idx / (H_ * 192));
  int rem = (int)(idx % (H_ * 192));
  int h = rem / 192, c = rem % 192;
  const float* src = q + (long)t * (H_ * 192) + h * 192;
  if (c < NOPE_) {
    qn[((long)t * H_ + h) * NOPE_ + c] = f2bf(src[c]);
  } else {
    int j = c - NOPE_;
    const float* pe = src + NOPE_;
    float x = pe[j];
    float rot = (j < 32) ? -pe[j + 32] : pe[j - 32];
    qpe[((long)t * H_ + h) * ROPE_ + j] =
        f2bf(x * cosb[t * ROPE_ + j] + rot * sinb[t * ROPE_ + j]);
  }
}

// kv (T,576) f32 -> c_kv bf16 (T,512) + transposed (512,T), roped k_pe (T,64)
__global__ void kv_post_k(const float* __restrict__ kv, const float* __restrict__ g,
                          const float* __restrict__ cosb, const float* __restrict__ sinb,
                          u16* __restrict__ ckv, u16* __restrict__ ckvt, u16* __restrict__ kpe) {
  __shared__ float red[256];
  int t = blockIdx.x, tid = threadIdx.x;
  const float* row = kv + (long)t * 576;
  float ss = 0.f;
  for (int i = tid; i < KVLR_; i += 256) { float v = row[i]; ss += v * v; }
  red[tid] = ss; __syncthreads();
  for (int o = 128; o > 0; o >>= 1) { if (tid < o) red[tid] += red[tid + o]; __syncthreads(); }
  float inv = rsqrtf(red[0] / (float)KVLR_ + 1e-6f);
  for (int i = tid; i < KVLR_; i += 256) {
    u16 b = f2bf(row[i] * inv * g[i]);
    ckv[(long)t * KVLR_ + i] = b;
    ckvt[(long)i * T_ + t] = b;
  }
  if (tid < ROPE_) {
    int j = tid;
    float x = row[KVLR_ + j];
    float rot = (j < 32) ? -row[KVLR_ + j + 32] : row[KVLR_ + j - 32];
    kpe[(long)t * ROPE_ + j] = f2bf(x * cosb[t * ROPE_ + j] + rot * sinb[t * ROPE_ + j]);
  }
}

// qi (T,HI,128) f32: rope first 64 dims, convert bf16
__global__ void qi_rope_k(const float* __restrict__ qi, const float* __restrict__ cosb,
                          const float* __restrict__ sinb, u16* __restrict__ out) {
  long idx = (long)blockIdx.x * blockDim.x + threadIdx.x;
  if (idx >= (long)T_ * HI_ * IDXD_) return;
  int t = (int)(idx / (HI_ * IDXD_));
  int rem = (int)(idx % (HI_ * IDXD_));
  int h = rem / IDXD_, c = rem % IDXD_;
  const float* src = qi + (long)t * (HI_ * IDXD_) + h * IDXD_;
  float v;
  if (c < ROPE_) {
    float x = src[c];
    float rot = (c < 32) ? -src[c + 32] : src[c - 32];
    v = x * cosb[t * ROPE_ + c] + rot * sinb[t * ROPE_ + c];
  } else v = src[c];
  out[idx] = f2bf(v);
}

// ki (T,128): layernorm + rope first 64, bf16 out. blockDim = 128.
__global__ void ki_post_k(const float* __restrict__ ki, const float* __restrict__ g,
                          const float* __restrict__ bta, const float* __restrict__ cosb,
                          const float* __restrict__ sinb, u16* __restrict__ out) {
  __shared__ float red[128];
  __shared__ float vals[128];
  int t = blockIdx.x, tid = threadIdx.x;
  float v = ki[(long)t * IDXD_ + tid];
  red[tid] = v; __syncthreads();
  for (int o = 64; o > 0; o >>= 1) { if (tid < o) red[tid] += red[tid + o]; __syncthreads(); }
  float mean = red[0] / 128.f;
  __syncthreads();
  float d = v - mean;
  red[tid] = d * d; __syncthreads();
  for (int o = 64; o > 0; o >>= 1) { if (tid < o) red[tid] += red[tid + o]; __syncthreads(); }
  float inv = rsqrtf(red[0] / 128.f + 1e-6f);
  float nv = d * inv * g[tid] + bta[tid];
  vals[tid] = nv; __syncthreads();
  float o;
  if (tid < ROPE_) {
    float rot = (tid < 32) ? -vals[tid + 32] : vals[tid - 32];
    o = nv * cosb[t * ROPE_ + tid] + rot * sinb[t * ROPE_ + tid];
  } else o = nv;
  out[(long)t * IDXD_ + tid] = f2bf(o);
}

// wgt = x @ Wiw^T + biw  (T x 16, tiny)
__global__ void wgt_k(const float* __restrict__ x, const float* __restrict__ Wiw,
                      const float* __restrict__ biw, float* __restrict__ wgt) {
  int idx = blockIdx.x * blockDim.x + threadIdx.x;
  if (idx >= T_ * HI_) return;
  int t = idx / HI_, h = idx % HI_;
  const float* xr = x + (long)t * HID_;
  const float* wr = Wiw + (long)h * HID_;
  float s = 0.f;
  for (int k = 0; k < HID_; ++k) s += xr[k] * wr[k];
  wgt[idx] = s + biw[h];
}

// ---------------------------------------------------------------------------
// Indexer scores (WMMA): iscore[t,s] = sum_h relu(qi_th . ki_s)/sqrt(128)*wgt
// ---------------------------------------------------------------------------
__global__ void indexer_scores_k(const u16* __restrict__ qi, const u16* __restrict__ ki,
                                 const float* __restrict__ wgt, float* __restrict__ iscore) {
  int wave = (int)((blockIdx.x * blockDim.x + threadIdx.x) >> 5);
  const int NT = T_ >> 4;
  if (wave >= NT * NT) return;
  int tm = wave / NT, tn = wave % NT;
  int m0 = tm << 4, s0 = tn << 4;
  int lane = threadIdx.x & 31;
  int n = s0 + (lane & 15), mh = (lane >> 4) << 3;
  if (s0 > m0) {  // tile entirely above causal diagonal
#pragma unroll
    for (int r = 0; r < 8; ++r) iscore[(long)(m0 + mh + r) * T_ + n] = NEGF;
    return;
  }
  float accr[8];
#pragma unroll
  for (int r = 0; r < 8; ++r) accr[r] = 0.f;
  for (int h = 0; h < HI_; ++h) {
    v8f d = {};
#pragma unroll
    for (int kc = 0; kc < IDXD_; kc += 32)
      d = wmma_bf16(load_fragA(qi + h * IDXD_, (long)HI_ * IDXD_, m0, kc),
                    load_fragB(ki, (long)IDXD_, s0, kc), d);
#pragma unroll
    for (int r = 0; r < 8; ++r) {
      int m = m0 + mh + r;
      accr[r] += fmaxf(d[r], 0.f) * IDX_SCALE * wgt[m * HI_ + h];
    }
  }
#pragma unroll
  for (int r = 0; r < 8; ++r) {
    int m = m0 + mh + r;
    iscore[(long)m * T_ + n] = (n <= m) ? accr[r] : NEGF;
  }
}

// Per-row top-k threshold via in-LDS bitonic sort (2048 f32, 256 threads)
__global__ void topk_threshold_k(const float* __restrict__ iscore, float* __restrict__ thr) {
  __shared__ float s[T_];
  int t = blockIdx.x, tid = threadIdx.x;
  for (int i = tid; i < T_; i += 256) s[i] = iscore[(long)t * T_ + i];
  __syncthreads();
  for (int k = 2; k <= T_; k <<= 1) {
    for (int j = k >> 1; j > 0; j >>= 1) {
      for (int i = tid; i < (T_ >> 1); i += 256) {
        int idx = ((i & ~(j - 1)) << 1) | (i & (j - 1));
        int ixj = idx | j;
        bool up = ((idx & k) == 0);
        float a = s[idx], b = s[ixj];
        if ((a > b) == up) { s[idx] = b; s[ixj] = a; }
      }
      __syncthreads();
    }
  }
  if (tid == 0) thr[t] = s[T_ - 1024];  // 1024th largest (ascending sort)
}

__global__ void build_mask_k(const float* __restrict__ iscore, const float* __restrict__ thr,
                             unsigned char* __restrict__ mask) {
  long idx = (long)blockIdx.x * blockDim.x + threadIdx.x;
  if (idx >= (long)T_ * T_) return;
  int m = (int)(idx / T_), sI = (int)(idx % T_);
  mask[idx] = (sI <= m && iscore[idx] >= thr[m]) ? (unsigned char)1 : (unsigned char)0;
}

// ---------------------------------------------------------------------------
// Attention scores (WMMA): pass1 = masked row max/sum, pass2 = write bf16 p
// ---------------------------------------------------------------------------
__device__ __forceinline__ v8f score_tile(const u16* qabs, const u16* qpe,
                                          const u16* ckv, const u16* kpe,
                                          int h, int m0, int s0) {
  v8f sc = {};
  for (int kc = 0; kc < KVLR_; kc += 32)
    sc = wmma_bf16(load_fragA(qabs + h * KVLR_, (long)H_ * KVLR_, m0, kc),
                   load_fragB(ckv, (long)KVLR_, s0, kc), sc);
#pragma unroll
  for (int kc = 0; kc < ROPE_; kc += 32)
    sc = wmma_bf16(load_fragA(qpe + h * ROPE_, (long)H_ * ROPE_, m0, kc),
                   load_fragB(kpe, (long)ROPE_, s0, kc), sc);
  return sc;
}

__global__ void attn_pass1_k(const u16* __restrict__ qabs, const u16* __restrict__ qpe,
                             const u16* __restrict__ ckv, const u16* __restrict__ kpe,
                             const unsigned char* __restrict__ mask,
                             float* __restrict__ rmax, float* __restrict__ rsum) {
  int wave = (int)((blockIdx.x * blockDim.x + threadIdx.x) >> 5);
  if (wave >= (T_ >> 4) * H_) return;
  int tm = wave >> 4, h = wave & 15;
  int m0 = tm << 4;
  int lane = threadIdx.x & 31, mh = (lane >> 4) << 3;
  float lmax[8], lsum[8];
#pragma unroll
  for (int r = 0; r < 8; ++r) { lmax[r] = NEGF; lsum[r] = 0.f; }
  for (int s0 = 0; s0 <= m0; s0 += 16) {
    v8f sc = score_tile(qabs, qpe, ckv, kpe, h, m0, s0);
    int n = s0 + (lane & 15);
#pragma unroll
    for (int r = 0; r < 8; ++r) {
      int m = m0 + mh + r;
      float v = sc[r] * ATT_SCALE;
      bool sel = (n <= m) && (mask[(long)m * T_ + n] != 0);
      if (sel) {
        if (v > lmax[r]) { lsum[r] = lsum[r] * __expf(lmax[r] - v) + 1.f; lmax[r] = v; }
        else lsum[r] += __expf(v - lmax[r]);
      }
    }
  }
#pragma unroll
  for (int off = 1; off < 16; off <<= 1) {
#pragma unroll
    for (int r = 0; r < 8; ++r) {
      float om = __shfl_xor(lmax[r], off, 32);
      float os = __shfl_xor(lsum[r], off, 32);
      float nm = fmaxf(lmax[r], om);
      float ns = lsum[r] * __expf(lmax[r] - nm) + os * __expf(om - nm);
      lmax[r] = nm; lsum[r] = ns;
    }
  }
  if ((lane & 15) == 0) {
#pragma unroll
    for (int r = 0; r < 8; ++r) {
      int m = m0 + mh + r;
      rmax[m * H_ + h] = lmax[r];
      rsum[m * H_ + h] = lsum[r];
    }
  }
}

__global__ void attn_pass2_k(const u16* __restrict__ qabs, const u16* __restrict__ qpe,
                             const u16* __restrict__ ckv, const u16* __restrict__ kpe,
                             const unsigned char* __restrict__ mask,
                             const float* __restrict__ rmax, const float* __restrict__ rsum,
                             u16* __restrict__ p) {
  int wave = (int)((blockIdx.x * blockDim.x + threadIdx.x) >> 5);
  if (wave >= (T_ >> 4) * H_) return;
  int tm = wave >> 4, h = wave & 15;
  int m0 = tm << 4;
  int lane = threadIdx.x & 31, mh = (lane >> 4) << 3;
  for (int s0 = 0; s0 < T_; s0 += 16) {
    int n = s0 + (lane & 15);
    if (s0 <= m0) {
      v8f sc = score_tile(qabs, qpe, ckv, kpe, h, m0, s0);
#pragma unroll
      for (int r = 0; r < 8; ++r) {
        int m = m0 + mh + r;
        float v = sc[r] * ATT_SCALE;
        bool sel = (n <= m) && (mask[(long)m * T_ + n] != 0);
        float pv = sel ? (__expf(v - rmax[m * H_ + h]) / rsum[m * H_ + h]) : 0.f;
        p[((long)m * H_ + h) * T_ + n] = f2bf(pv);
      }
    } else {
#pragma unroll
      for (int r = 0; r < 8; ++r)
        p[((long)(m0 + mh + r) * H_ + h) * T_ + n] = (u16)0;
    }
  }
}

// ---------------------------------------------------------------------------
// Host orchestration
// ---------------------------------------------------------------------------
extern "C" void kernel_launch(void* const* d_in, const int* in_sizes, int n_in,
                              void* d_out, int out_size, void* d_ws, size_t ws_size,
                              hipStream_t stream) {
  (void)in_sizes; (void)n_in; (void)out_size; (void)ws_size;
  const float* x     = (const float*)d_in[0];
  const float* cosb  = (const float*)d_in[1];
  const float* sinb  = (const float*)d_in[2];
  const float* Wq_a  = (const float*)d_in[3];
  const float* g_qa  = (const float*)d_in[4];
  const float* Wq_b  = (const float*)d_in[5];
  const float* Wkv_a = (const float*)d_in[6];
  const float* g_kva = (const float*)d_in[7];
  const float* Wkv_b = (const float*)d_in[8];
  const float* Wo    = (const float*)d_in[9];
  const float* Wiq   = (const float*)d_in[10];
  const float* Wik   = (const float*)d_in[11];
  const float* ikG   = (const float*)d_in[12];
  const float* ikB   = (const float*)d_in[13];
  const float* Wiw   = (const float*)d_in[14];
  const float* biw   = (const float*)d_in[15];
  float* outp = (float*)d_out;

  char* wsb = (char*)d_ws;
  size_t off = 0;
  auto alloc = [&](size_t bytes) -> void* {
    void* p = wsb + off;
    off = (off + bytes + 255) & ~(size_t)255;
    return p;
  };

  u16* xb     = (u16*)alloc((size_t)T_ * HID_ * 2);
  u16* wqa_b  = (u16*)alloc((size_t)QLR_ * HID_ * 2);
  u16* wqb_b  = (u16*)alloc((size_t)(H_ * 192) * QLR_ * 2);
  u16* wkva_b = (u16*)alloc((size_t)576 * HID_ * 2);
  u16* wukt_b = (u16*)alloc((size_t)H_ * KVLR_ * NOPE_ * 2);
  u16* wuv_b  = (u16*)alloc((size_t)H_ * V_ * KVLR_ * 2);
  u16* wo_b   = (u16*)alloc((size_t)HID_ * (H_ * V_) * 2);
  u16* wiq_b  = (u16*)alloc((size_t)(HI_ * IDXD_) * QLR_ * 2);
  u16* wik_b  = (u16*)alloc((size_t)IDXD_ * HID_ * 2);
  float* qr_f = (float*)alloc((size_t)T_ * QLR_ * 4);
  u16* qr_b   = (u16*)alloc((size_t)T_ * QLR_ * 2);
  float* q_f  = (float*)alloc((size_t)T_ * H_ * 192 * 4);
  u16* qn_b   = (u16*)alloc((size_t)T_ * H_ * NOPE_ * 2);
  u16* qpe_b  = (u16*)alloc((size_t)T_ * H_ * ROPE_ * 2);
  float* kv_f = (float*)alloc((size_t)T_ * 576 * 4);
  u16* ckv_b  = (u16*)alloc((size_t)T_ * KVLR_ * 2);
  u16* ckvt_b = (u16*)alloc((size_t)KVLR_ * T_ * 2);
  u16* kpe_b  = (u16*)alloc((size_t)T_ * ROPE_ * 2);
  u16* qabs_b = (u16*)alloc((size_t)T_ * H_ * KVLR_ * 2);
  float* qi_f = (float*)alloc((size_t)T_ * HI_ * IDXD_ * 4);
  u16* qi_b   = (u16*)alloc((size_t)T_ * HI_ * IDXD_ * 2);
  float* ki_f = (float*)alloc((size_t)T_ * IDXD_ * 4);
  u16* ki_b   = (u16*)alloc((size_t)T_ * IDXD_ * 2);
  float* wgt_f = (float*)alloc((size_t)T_ * HI_ * 4);
  float* isc  = (float*)alloc((size_t)T_ * T_ * 4);
  float* thr  = (float*)alloc((size_t)T_ * 4);
  unsigned char* mk = (unsigned char*)alloc((size_t)T_ * T_);
  float* rmx  = (float*)alloc((size_t)T_ * H_ * 4);
  float* rsm  = (float*)alloc((size_t)T_ * H_ * 4);
  u16* p_b    = (u16*)alloc((size_t)T_ * H_ * T_ * 2);
  u16* olat_b = (u16*)alloc((size_t)T_ * H_ * KVLR_ * 2);
  u16* o_b    = (u16*)alloc((size_t)T_ * (H_ * V_) * 2);

  auto nblk = [](long n, int b) { return (int)((n + b - 1) / b); };
  auto cvt = [&](const float* in, u16* o, long n) {
    convert_bf16_k<<<2048, 256, 0, stream>>>(in, o, n);
  };
  auto gemm = [&](const u16* A, long lda, long sA, const u16* B, long ldb, long sB,
                  void* C, long ldc, long sC, bool cbf, int M, int N, int K, int bat) {
    long waves = (long)bat * (M / 32) * (N / 64);
    gemm_wmma<<<nblk(waves * 32, 256), 256, 0, stream>>>(
        A, lda, sA, B, ldb, sB, C, ldc, sC, cbf ? 1 : 0, M, N, K, bat);
  };

  // --- convert operands to bf16 ---
  cvt(x, xb, (long)T_ * HID_);
  cvt(Wq_a, wqa_b, (long)QLR_ * HID_);
  cvt(Wq_b, wqb_b, (long)(H_ * 192) * QLR_);
  cvt(Wkv_a, wkva_b, (long)576 * HID_);
  cvt(Wo, wo_b, (long)HID_ * H_ * V_);
  cvt(Wiq, wiq_b, (long)(HI_ * IDXD_) * QLR_);
  cvt(Wik, wik_b, (long)IDXD_ * HID_);
  split_wkvb_k<<<nblk(2L * H_ * KVLR_ * NOPE_, 256), 256, 0, stream>>>(Wkv_b, wukt_b, wuv_b);

  // --- q path ---
  gemm(xb, HID_, 0, wqa_b, HID_, 0, qr_f, QLR_, 0, false, T_, QLR_, HID_, 1);
  rmsnorm_rows_bf_k<<<T_, 256, 0, stream>>>(qr_f, g_qa, qr_b, QLR_);
  gemm(qr_b, QLR_, 0, wqb_b, QLR_, 0, q_f, H_ * 192, 0, false, T_, H_ * 192, QLR_, 1);
  extract_q_k<<<nblk((long)T_ * H_ * 192, 256), 256, 0, stream>>>(q_f, cosb, sinb, qn_b, qpe_b);

  // --- kv path ---
  gemm(xb, HID_, 0, wkva_b, HID_, 0, kv_f, 576, 0, false, T_, 576, HID_, 1);
  kv_post_k<<<T_, 256, 0, stream>>>(kv_f, g_kva, cosb, sinb, ckv_b, ckvt_b, kpe_b);

  // --- absorbed q: q_abs[t,h,:] = q_nope[t,h,:] @ W_UK[h] (batched over heads) ---
  gemm(qn_b, (long)H_ * NOPE_, NOPE_, wukt_b, NOPE_, (long)KVLR_ * NOPE_,
       qabs_b, (long)H_ * KVLR_, KVLR_, true, T_, KVLR_, NOPE_, H_);

  // --- indexer ---
  gemm(qr_b, QLR_, 0, wiq_b, QLR_, 0, qi_f, HI_ * IDXD_, 0, false, T_, HI_ * IDXD_, QLR_, 1);
  qi_rope_k<<<nblk((long)T_ * HI_ * IDXD_, 256), 256, 0, stream>>>(qi_f, cosb, sinb, qi_b);
  gemm(xb, HID_, 0, wik_b, HID_, 0, ki_f, IDXD_, 0, false, T_, IDXD_, HID_, 1);
  ki_post_k<<<T_, 128, 0, stream>>>(ki_f, ikG, ikB, cosb, sinb, ki_b);
  wgt_k<<<nblk((long)T_ * HI_, 256), 256, 0, stream>>>(x, Wiw, biw, wgt_f);
  indexer_scores_k<<<nblk((long)(T_ / 16) * (T_ / 16) * 32, 256), 256, 0, stream>>>(qi_b, ki_b, wgt_f, isc);
  topk_threshold_k<<<T_, 256, 0, stream>>>(isc, thr);
  build_mask_k<<<nblk((long)T_ * T_, 256), 256, 0, stream>>>(isc, thr, mk);

  // --- sparse attention (two-pass softmax over selected set) ---
  attn_pass1_k<<<nblk((long)(T_ / 16) * H_ * 32, 256), 256, 0, stream>>>(
      qabs_b, qpe_b, ckv_b, kpe_b, mk, rmx, rsm);
  attn_pass2_k<<<nblk((long)(T_ / 16) * H_ * 32, 256), 256, 0, stream>>>(
      qabs_b, qpe_b, ckv_b, kpe_b, mk, rmx, rsm, p_b);

  // --- o_lat = p @ c_kv (batched heads), o = o_lat @ W_UV^T, out = o @ Wo^T ---
  gemm(p_b, (long)H_ * T_, T_, ckvt_b, T_, 0,
       olat_b, (long)H_ * KVLR_, KVLR_, true, T_, KVLR_, T_, H_);
  gemm(olat_b, (long)H_ * KVLR_, KVLR_, wuv_b, KVLR_, (long)V_ * KVLR_,
       o_b, (long)H_ * V_, V_, true, T_, V_, KVLR_, H_);
  gemm(o_b, (long)H_ * V_, 0, wo_b, (long)H_ * V_, 0,
       outp, HID_, 0, false, T_, HID_, H_ * V_, 1);
}
